// MultiHeadAttention_12756052869442
// MI455X (gfx1250) — matrix operand order repaired and checked by reading, not compiled, not run on hardware
//
#include <hip/hip_runtime.h>
#include <hip/hip_bf16.h>

// ---- problem constants -----------------------------------------------------
#define EMD    768
#define HEADS  8
#define HDIM   96
#define BATCH  4
#define SEQ    2048
#define MROWS  (BATCH * SEQ)   // 8192 rows of x / out
#define NKT    (SEQ / 32)      // 64 k-tiles in attention

// ---- WMMA fragment types (CDNA5, wave32) ----------------------------------
typedef __bf16 v16bf __attribute__((ext_vector_type(16)));
typedef float  v8f   __attribute__((ext_vector_type(8)));

union FragAB {
    v16bf        bf;
    unsigned int u[8];
    uint4        q[2];
};

static __device__ __forceinline__ unsigned short f32_to_bf16_rne(float f) {
    unsigned int u = __float_as_uint(f);
    u += 0x7FFFu + ((u >> 16) & 1u);          // round-to-nearest-even
    return (unsigned short)(u >> 16);
}

// ---- CDNA5 async global->LDS staging (ASYNCcnt-tracked) --------------------
#if __has_builtin(__builtin_amdgcn_global_load_async_to_lds_b128)
#define HAVE_ASYNC_LDS 1
#else
#define HAVE_ASYNC_LDS 0
#endif

// builtin prototype (from hipcc diagnostic): (v4i as1*, v4i as3*, imm, imm)
typedef int v4i_t __attribute__((ext_vector_type(4)));
typedef __attribute__((address_space(1))) v4i_t* as1_v4i;
typedef __attribute__((address_space(3))) v4i_t* as3_v4i;

static __device__ __forceinline__ void wait_async_le12() {
#if HAVE_ASYNC_LDS
    asm volatile("s_wait_asynccnt 12" ::: "memory");
#endif
}
static __device__ __forceinline__ void wait_async_le0() {
#if HAVE_ASYNC_LDS
    asm volatile("s_wait_asynccnt 0" ::: "memory");
#endif
}

// stage one 32x96 bf16 V tile (6 KB) : 12 x b128 per lane (wave-cooperative)
static __device__ __forceinline__ void stage_v_tile(
    const unsigned short* __restrict__ src, unsigned short* dst_lds, int lane)
{
#if HAVE_ASYNC_LDS
#pragma unroll
    for (int i = 0; i < 12; ++i) {
        __builtin_amdgcn_global_load_async_to_lds_b128(
            (as1_v4i)(const_cast<unsigned short*>(src) + (size_t)(lane + 32 * i) * 8),
            (as3_v4i)(dst_lds + (size_t)(lane + 32 * i) * 8),
            0, 0);
    }
#else
    const uint4* s = (const uint4*)src;
    uint4*       d = (uint4*)dst_lds;
#pragma unroll
    for (int i = 0; i < 12; ++i) d[lane + 32 * i] = s[lane + 32 * i];
#endif
}

// ---- kernel 0: f32 -> bf16 conversion -------------------------------------
__global__ __launch_bounds__(256) void cvt_f32_to_bf16(
    const float* __restrict__ src, unsigned short* __restrict__ dst, int n)
{
    int i = blockIdx.x * blockDim.x + threadIdx.x;
    if (i < n) dst[i] = f32_to_bf16_rne(src[i]);
}

// ---- kernel 1/3: bf16 GEMM  out[m,o] = sum_k A[m,k]*W[o,k] + bias[o] ------
__global__ __launch_bounds__(256) void gemm_bf16_wmma(
    const unsigned short* __restrict__ A,
    const unsigned short* __restrict__ W,
    const float*          __restrict__ bias,
    unsigned short*       __restrict__ Obf,
    float*                __restrict__ Of32,
    int split_heads)
{
    const int wave = threadIdx.x >> 5;
    const int lane = threadIdx.x & 31;
    const int lhi  = lane >> 4;           // lane group (0/1)
    const int l15  = lane & 15;

    const int M0 = blockIdx.y * 32 + (wave >> 2) * 16;
    const int N0 = blockIdx.x * 64 + (wave & 3) * 16;

    v8f c = {0.f, 0.f, 0.f, 0.f, 0.f, 0.f, 0.f, 0.f};

    const unsigned short* arow = A + (size_t)(M0 + l15) * EMD;  // A: M = l15
    const unsigned short* brow = W + (size_t)(N0 + l15) * EMD;  // B: N = l15

    for (int kk = 0; kk < EMD; kk += 32) {
        FragAB a, b;
        // A 16x32 bf16: lanes0-15 K in {0..7,16..23}; lanes16-31 K in {8..15,24..31}
        a.q[0] = *(const uint4*)(arow + kk + 8 * lhi);
        a.q[1] = *(const uint4*)(arow + kk + 8 * lhi + 16);
        // B 32x16 bf16: lanes0-15 K in {0..7,16..23}; lanes16-31 K in {8..15,24..31}
        b.q[0] = *(const uint4*)(brow + kk + 8 * lhi);
        b.q[1] = *(const uint4*)(brow + kk + 8 * lhi + 16);
        c = __builtin_amdgcn_wmma_f32_16x16x32_bf16(false, a.bf, false, b.bf,
                                                    (short)0, c, false, false);
    }

    const int   o   = N0 + l15;
    const float bia = bias[o];
    const int   h   = o / HDIM;
    const int   d   = o % HDIM;

#pragma unroll
    for (int v = 0; v < 8; ++v) {
        const int   m   = M0 + v + 8 * lhi;     // C: M = v + 8*laneGroup
        const float val = c[v] + bia;
        if (split_heads) {
            const int bb = m >> 11, nn = m & 2047;
            Obf[(((size_t)(bb * HEADS + h)) * SEQ + nn) * HDIM + d] =
                f32_to_bf16_rne(val);
        } else {
            Of32[(size_t)m * EMD + o] = val;
        }
    }
}

// ---- kernel 2: fused attention with softmax over HEADS --------------------
// grid = (SEQ/16, BATCH); block = 256 threads = 8 waves; wave h owns head h.
__global__ __launch_bounds__(256) void attn_headsoftmax_wmma(
    const unsigned short* __restrict__ Qm,   // [B,H,N,D] bf16
    const unsigned short* __restrict__ Km,
    const unsigned short* __restrict__ Vm,
    unsigned short*       __restrict__ Om)   // [B,N,E] bf16
{
    const int b    = blockIdx.y;
    const int q0   = blockIdx.x << 4;
    const int h    = threadIdx.x >> 5;
    const int lane = threadIdx.x & 31;
    const int lhi  = lane >> 4;
    const int l15  = lane & 15;

    __shared__ __align__(16) float          lds_e[8][2][8][32];     // energy, 16 KB
    __shared__ __align__(4)  unsigned short lds_a[8][16][32];       // attn bf16, 8 KB
    __shared__ __align__(16) unsigned short lds_v[8][2][32][96];    // V dbl-buf, 96 KB

    const size_t headBase = (size_t)(b * HEADS + h) * SEQ;

    // Q A-fragments for this head's 16 q rows: 3 chunks of K=32 over d=96
    FragAB qa[3];
    {
        const unsigned short* qrow = Qm + (headBase + q0 + l15) * HDIM;
#pragma unroll
        for (int c = 0; c < 3; ++c) {
            const int base = 32 * c + 8 * lhi;
            qa[c].q[0] = *(const uint4*)(qrow + base);
            qa[c].q[1] = *(const uint4*)(qrow + base + 16);
        }
    }

    const v8f vzero = {0.f, 0.f, 0.f, 0.f, 0.f, 0.f, 0.f, 0.f};
    v8f acc[6];
#pragma unroll
    for (int t = 0; t < 6; ++t) acc[t] = vzero;

    const float inv_sqrt_emd = 0.036084391824351615f;  // 1/sqrt(768)

    // prologue: stage V tile 0 into buffer 0 (async on CDNA5)
    stage_v_tile(Vm + headBase * HDIM, &lds_v[h][0][0][0], lane);

    for (int kt = 0; kt < NKT; ++kt) {
        const int k0  = kt << 5;
        const int cur = kt & 1;

        // stage next V tile into the other buffer while we compute
        if (kt + 1 < NKT)
            stage_v_tile(Vm + (headBase + k0 + 32) * HDIM,
                         &lds_v[h][cur ^ 1][0][0], lane);

        // energy tiles: e0 = q-tile x K[k0..k0+15]^T, e1 = x K[k0+16..k0+31]^T
        const unsigned short* kbase = Km + (headBase + k0) * HDIM;
        v8f e0 = vzero, e1 = vzero;
#pragma unroll
        for (int c = 0; c < 3; ++c) {
            FragAB kb0, kb1;
            const unsigned short* kr0 = kbase + (size_t)l15 * HDIM + 32 * c + 8 * lhi;
            const unsigned short* kr1 = kr0 + 16 * HDIM;
            kb0.q[0] = *(const uint4*)(kr0);
            kb0.q[1] = *(const uint4*)(kr0 + 16);
            kb1.q[0] = *(const uint4*)(kr1);
            kb1.q[1] = *(const uint4*)(kr1 + 16);
            e0 = __builtin_amdgcn_wmma_f32_16x16x32_bf16(false, qa[c].bf, false, kb0.bf,
                                                         (short)0, e0, false, false);
            e1 = __builtin_amdgcn_wmma_f32_16x16x32_bf16(false, qa[c].bf, false, kb1.bf,
                                                         (short)0, e1, false, false);
        }

        if (kt + 1 < NKT)   // prefetch next K tile -> global_prefetch_b8
            __builtin_prefetch((const void*)(kbase + 32 * HDIM), 0, 3);

        // publish energies so every wave can see all 8 heads at its slot
#pragma unroll
        for (int v = 0; v < 8; ++v) {
            lds_e[h][0][v][lane] = e0[v];
            lds_e[h][1][v][lane] = e1[v];
        }
        __syncthreads();

        // softmax over heads (local to each (q,k)!) + 1/sqrt(EMD) scale
#pragma unroll
        for (int t = 0; t < 2; ++t) {
#pragma unroll
            for (int v = 0; v < 8; ++v) {
                float vals[8], m = -3.4e38f;
#pragma unroll
                for (int hh = 0; hh < 8; ++hh) {
                    vals[hh] = lds_e[hh][t][v][lane];
                    m = fmaxf(m, vals[hh]);
                }
                float s = 0.f;
#pragma unroll
                for (int hh = 0; hh < 8; ++hh) s += __expf(vals[hh] - m);
                const float mine = __expf(vals[h] - m) * (inv_sqrt_emd / s);
                const int qq = v + 8 * lhi;     // C layout: M = v + 8*laneGroup
                const int kk = 16 * t + l15;    //            N = l15
                lds_a[h][qq][kk] = f32_to_bf16_rne(mine);
            }
        }
        __syncthreads();   // protects lds_e against next-iteration overwrite

        // repack attention probs into A-fragment layout (16 q x 32 k)
        FragAB af;
#pragma unroll
        for (int v = 0; v < 8; ++v) {
            const int kk = 2 * (v & 3) + 8 * lhi + 16 * (v >> 2);   // even
            af.u[v] = *(const unsigned int*)&lds_a[h][l15][kk];
        }

        // ensure current V tile's async loads have landed (in-order completion:
        // <=12 outstanding means only the *next* tile's loads remain in flight)
        if (kt + 1 < NKT) wait_async_le12();
        else              wait_async_le0();

        // out += attn * V  (6 tiles of 16 d-columns)
#pragma unroll
        for (int t2 = 0; t2 < 6; ++t2) {
            FragAB vbf;
#pragma unroll
            for (int v = 0; v < 8; ++v) {
                const int kk = 2 * (v & 3) + 8 * lhi + 16 * (v >> 2);
                const unsigned int lo = lds_v[h][cur][kk][16 * t2 + l15];
                const unsigned int hi = lds_v[h][cur][kk + 1][16 * t2 + l15];
                vbf.u[v] = lo | (hi << 16);
            }
            acc[t2] = __builtin_amdgcn_wmma_f32_16x16x32_bf16(false, af.bf, false, vbf.bf,
                                                              (short)0, acc[t2], false, false);
        }
    }

    // merged-head bf16 output [b, n, h*96 + d]
#pragma unroll
    for (int t2 = 0; t2 < 6; ++t2) {
#pragma unroll
        for (int v = 0; v < 8; ++v) {
            const int qq = v + 8 * lhi;
            const int dd = 16 * t2 + l15;
            Om[((size_t)b * SEQ + q0 + qq) * EMD + h * HDIM + dd] =
                f32_to_bf16_rne(acc[t2][v]);
        }
    }
}

// ---- host-side orchestration ----------------------------------------------
extern "C" void kernel_launch(void* const* d_in, const int* in_sizes, int n_in,
                              void* d_out, int out_size, void* d_ws, size_t ws_size,
                              hipStream_t stream) {
    (void)in_sizes; (void)n_in; (void)out_size; (void)ws_size;

    const float* x  = (const float*)d_in[0];
    const float* Wq = (const float*)d_in[1];
    const float* bq = (const float*)d_in[2];
    const float* Wk = (const float*)d_in[3];
    const float* bk = (const float*)d_in[4];
    const float* Wv = (const float*)d_in[5];
    const float* bv = (const float*)d_in[6];
    const float* Wo = (const float*)d_in[7];
    const float* bo = (const float*)d_in[8];
    float* out = (float*)d_out;

    const size_t nX = (size_t)MROWS * EMD;   // 6,291,456
    const size_t nW = (size_t)EMD * EMD;     //   589,824

    unsigned short* w   = (unsigned short*)d_ws;
    unsigned short* xb  = w;  w += nX;
    unsigned short* wqb = w;  w += nW;
    unsigned short* wkb = w;  w += nW;
    unsigned short* wvb = w;  w += nW;
    unsigned short* wob = w;  w += nW;
    unsigned short* qb  = w;  w += nX;
    unsigned short* kb  = w;  w += nX;
    unsigned short* vb  = w;  w += nX;
    unsigned short* ob  = w;  w += nX;

    // 0) convert inputs to bf16
    cvt_f32_to_bf16<<<dim3((unsigned)((nX + 255) / 256)), 256, 0, stream>>>(x,  xb,  (int)nX);
    cvt_f32_to_bf16<<<dim3((unsigned)((nW + 255) / 256)), 256, 0, stream>>>(Wq, wqb, (int)nW);
    cvt_f32_to_bf16<<<dim3((unsigned)((nW + 255) / 256)), 256, 0, stream>>>(Wk, wkb, (int)nW);
    cvt_f32_to_bf16<<<dim3((unsigned)((nW + 255) / 256)), 256, 0, stream>>>(Wv, wvb, (int)nW);
    cvt_f32_to_bf16<<<dim3((unsigned)((nW + 255) / 256)), 256, 0, stream>>>(Wo, wob, (int)nW);

    // 1) Q/K/V projections -> [b,h,n,d] bf16
    const dim3 ggrid(EMD / 64, MROWS / 32);   // (12, 256)
    gemm_bf16_wmma<<<ggrid, 256, 0, stream>>>(xb, wqb, bq, qb, nullptr, 1);
    gemm_bf16_wmma<<<ggrid, 256, 0, stream>>>(xb, wkb, bk, kb, nullptr, 1);
    gemm_bf16_wmma<<<ggrid, 256, 0, stream>>>(xb, wvb, bv, vb, nullptr, 1);

    // 2) fused attention (softmax over heads), streamed over k
    attn_headsoftmax_wmma<<<dim3(SEQ / 16, BATCH), 256, 0, stream>>>(qb, kb, vb, ob);

    // 3) output projection -> f32 d_out
    gemm_bf16_wmma<<<ggrid, 256, 0, stream>>>(ob, wob, bo, nullptr, out, 0);
}
